// HypHawkes_70179765617222
// MI455X (gfx1250) — compile-verified, matrix operand order checked by reading, not run on hardware
//
#include <hip/hip_runtime.h>
#include <hip/hip_bf16.h>
#include <math.h>

typedef __attribute__((ext_vector_type(16))) _Float16 v16h;
typedef __attribute__((ext_vector_type(8)))  _Float16 v8h;
typedef __attribute__((ext_vector_type(8)))  float    v8f;

// 16-byte chunk type matching the async-load builtin's expected pointee.
typedef int v4i __attribute__((vector_size(16)));
typedef __attribute__((address_space(1))) v4i as1_v4i;   // global
typedef __attribute__((address_space(3))) v4i as3_v4i;   // LDS

// ---------------------------------------------------------------------------
// CDNA5 async global->LDS copy (ASYNCcnt-tracked).  Falls back to a
// synchronous global load + ds_store pair if the builtin is unavailable.
// ---------------------------------------------------------------------------
#if __has_builtin(__builtin_amdgcn_global_load_async_to_lds_b128)
#define USE_ASYNC_LDS 1
#else
#define USE_ASYNC_LDS 0
#endif

__device__ __forceinline__ void copy16_g2l(const _Float16* g, _Float16* l) {
#if USE_ASYNC_LDS
    __builtin_amdgcn_global_load_async_to_lds_b128(
        (as1_v4i*)g, (as3_v4i*)l, 0, 0);
#else
    *(v8h*)l = *(const v8h*)g;
#endif
}

__device__ __forceinline__ void wait_async_copies() {
#if USE_ASYNC_LDS
#if __has_builtin(__builtin_amdgcn_s_wait_asynccnt)
    __builtin_amdgcn_s_wait_asynccnt(0);
#else
    asm volatile("s_wait_asynccnt 0" ::: "memory");
#endif
#endif
}

// ---------------------------------------------------------------------------
// LDS tiling parameters.
// K-step 32 halves per stage; row stride padded 32 -> 40 halves (80 B) so the
// 16 row-lanes of a ds_load_b128 fragment read land in disjoint bank groups.
// A tile: 128 rows, B tile: 256 rows, double buffered  ->  60 KB total LDS.
// ---------------------------------------------------------------------------
#define KSTEP      32
#define LDS_STRIDE 40
#define A_ROWS     128
#define B_ROWS     256

// Stage one (A,B) k-slab into LDS buffer `buf` (async when available).
__device__ __forceinline__ void stage_tiles(const _Float16* __restrict__ gA,
                                            const _Float16* __restrict__ gB,
                                            _Float16* sA, _Float16* sB,
                                            int buf, int kBase, int K, int tid) {
    _Float16* aDst = sA + buf * (A_ROWS * LDS_STRIDE);
    _Float16* bDst = sB + buf * (B_ROWS * LDS_STRIDE);
    const _Float16* aSrc = gA + kBase;
    const _Float16* bSrc = gB + kBase;
#pragma unroll
    for (int c = 0; c < 2; ++c) {                 // A: 128 rows * 4 chunks = 512
        const int ch  = tid + c * 256;
        const int row = ch >> 2;
        const int kc  = ch & 3;
        copy16_g2l(aSrc + (size_t)row * K + kc * 8,
                   aDst + row * LDS_STRIDE + kc * 8);
    }
#pragma unroll
    for (int c = 0; c < 4; ++c) {                 // B: 256 rows * 4 chunks = 1024
        const int ch  = tid + c * 256;
        const int row = ch >> 2;
        const int kc  = ch & 3;
        copy16_g2l(bSrc + (size_t)row * K + kc * 8,
                   bDst + row * LDS_STRIDE + kc * 8);
    }
}

// Load one 16x32 f16 WMMA operand fragment from a padded LDS tile.
// CDNA5 16-bit operand layout: lanes 0-15 hold row=lane, K=0..7 / 16..23;
// lanes 16-31 hold K=8..15 / 24..31.
__device__ __forceinline__ v16h frag_from_lds(const _Float16* tile, int rowBase,
                                              int lane) {
    const int r    = rowBase + (lane & 15);
    const int ksel = (lane >> 4) << 3;            // 0 or 8
    const _Float16* p = tile + r * LDS_STRIDE + ksel;
    v8h lo = *(const v8h*)p;                      // K = ksel .. ksel+7
    v8h hi = *(const v8h*)(p + 16);               // K = ksel+16 .. ksel+23
    return __builtin_shufflevector(lo, hi, 0, 1, 2, 3, 4, 5, 6, 7,
                                           8, 9, 10, 11, 12, 13, 14, 15);
}

// ---------------------------------------------------------------------------
// NT GEMM: C[i,j] = sum_k A[i,k]*B[j,k];  A:[M,K] f16, B:[N,K] f16.
// 256 threads = 8 waves (2 M x 4 N), wave tile 64x64 (4x4 WMMA, f32 accum),
// workgroup tile 128x256, double-buffered async LDS staging over K.
// ---------------------------------------------------------------------------
template <bool STORE_HALF>
__global__ __launch_bounds__(256)
void gemm_nt_wmma(const _Float16* __restrict__ A, const _Float16* __restrict__ B,
                  float* __restrict__ Cf, _Float16* __restrict__ Ch,
                  int M, int N, int K) {
    __shared__ __attribute__((aligned(16))) _Float16 sA[2 * A_ROWS * LDS_STRIDE];
    __shared__ __attribute__((aligned(16))) _Float16 sB[2 * B_ROWS * LDS_STRIDE];

    const int tid     = threadIdx.x;
    const int lane    = tid & 31;
    const int wave    = tid >> 5;
    const int waveRow = wave >> 2;                // 0..1
    const int waveCol = wave & 3;                 // 0..3
    const int mTile   = blockIdx.y * A_ROWS;
    const int nTile   = blockIdx.x * B_ROWS;

    const _Float16* gA = A + (size_t)mTile * K;
    const _Float16* gB = B + (size_t)nTile * K;

    v8f acc[4][4] = {};

    stage_tiles(gA, gB, sA, sB, 0, 0, K, tid);

    const int nK = K / KSTEP;
    for (int kt = 0; kt < nK; ++kt) {
        wait_async_copies();          // my staging from previous iter complete
        __syncthreads();              // everyone's staging complete; all reads
                                      // of the other buffer are done too
        const int cur = kt & 1;
        if (kt + 1 < nK)
            stage_tiles(gA, gB, sA, sB, cur ^ 1, (kt + 1) * KSTEP, K, tid);

        const _Float16* aBuf = sA + cur * (A_ROWS * LDS_STRIDE);
        const _Float16* bBuf = sB + cur * (B_ROWS * LDS_STRIDE);

        v16h af[4], bf[4];
#pragma unroll
        for (int i = 0; i < 4; ++i)
            af[i] = frag_from_lds(aBuf, waveRow * 64 + 16 * i, lane);
#pragma unroll
        for (int j = 0; j < 4; ++j)
            bf[j] = frag_from_lds(bBuf, waveCol * 64 + 16 * j, lane);
#pragma unroll
        for (int i = 0; i < 4; ++i)
#pragma unroll
            for (int j = 0; j < 4; ++j)
                acc[i][j] = __builtin_amdgcn_wmma_f32_16x16x32_f16(
                    false, af[i], false, bf[j], (short)0, acc[i][j], false, false);
    }

    // C/D layout: lane = N + 16*(M>=8); VGPR v holds M = v (+8 for high lanes)
    const int mBase = mTile + waveRow * 64;
    const int nBase = nTile + waveCol * 64;
    const int rOff  = (lane >> 4) << 3;           // 0 or 8
    const int c0    = lane & 15;
#pragma unroll
    for (int i = 0; i < 4; ++i) {
#pragma unroll
        for (int j = 0; j < 4; ++j) {
            const int col = nBase + 16 * j + c0;
#pragma unroll
            for (int v = 0; v < 8; ++v) {
                const int row = mBase + 16 * i + rOff + v;
                if (STORE_HALF)
                    Ch[(size_t)row * N + col] = (_Float16)acc[i][j][v];
                else
                    Cf[(size_t)row * N + col] = acc[i][j][v];
            }
        }
    }
}

// ---------------------------------------------------------------------------
// Row-wise expmap0 + project, f32 -> f16.  One workgroup per row (d == 1024,
// 256 threads x 4 elements).  Rows [0, nq) come from qin, rest from cin.
// ---------------------------------------------------------------------------
__global__ __launch_bounds__(256)
void prep_expmap_rows(const float* __restrict__ qin, const float* __restrict__ cin,
                      _Float16* __restrict__ qh, _Float16* __restrict__ ch,
                      const float* __restrict__ curv, int d, int nq) {
    __shared__ float red[256];
    const int row = blockIdx.x;
    const float* src;
    _Float16* dst;
    if (row < nq) { src = qin + (size_t)row * d;        dst = qh + (size_t)row * d; }
    else          { src = cin + (size_t)(row - nq) * d; dst = ch + (size_t)(row - nq) * d; }

    const int tid = threadIdx.x;
    float x[4];
    float ss = 0.0f;
#pragma unroll
    for (int t = 0; t < 4; ++t) { x[t] = src[tid + 256 * t]; ss += x[t] * x[t]; }
    red[tid] = ss;
    __syncthreads();
    for (int s = 128; s > 0; s >>= 1) {
        if (tid < s) red[tid] += red[tid + s];
        __syncthreads();
    }
    const float nrm = sqrtf(red[0]);
    const float sc  = sqrtf(curv[0]);
    const float n1  = fmaxf(nrm, 1e-5f);
    const float g   = tanhf(sc * n1) / (sc * n1);     // expmap0 scale
    const float yn  = fmaxf(nrm * g, 1e-5f);          // ||expmap0(u)||
    const float mxn = (1.0f - 4e-3f) / sc;            // projection radius
    const float f   = (yn > mxn) ? (mxn / yn) : 1.0f;
    const float s2  = g * f;
#pragma unroll
    for (int t = 0; t < 4; ++t) dst[tid + 256 * t] = (_Float16)(x[t] * s2);
}

__global__ __launch_bounds__(256)
void cvt_f32_f16(const float* __restrict__ in, _Float16* __restrict__ out, int n) {
    const int i = blockIdx.x * 256 + threadIdx.x;
    if (i < n) out[i] = (_Float16)in[i];
}

// ---------------------------------------------------------------------------
// In-place per-row: softmax(S/d) then expmap0 + project.  m == 4096:
// 256 threads x 16 values held in registers; 3 LDS tree reductions.
// ---------------------------------------------------------------------------
__global__ __launch_bounds__(256)
void softmax_expmap_rows(float* __restrict__ S, const float* __restrict__ curv,
                         int m, float invd) {
    __shared__ float red[256];
    const int row = blockIdx.x;
    float* p = S + (size_t)row * m;
    const int tid = threadIdx.x;

    float v[16];
    float mx = -3.0e38f;
#pragma unroll
    for (int t = 0; t < 16; ++t) {
        v[t] = p[tid + 256 * t] * invd;
        mx = fmaxf(mx, v[t]);
    }
    red[tid] = mx;
    __syncthreads();
    for (int s = 128; s > 0; s >>= 1) {
        if (tid < s) red[tid] = fmaxf(red[tid], red[tid + s]);
        __syncthreads();
    }
    mx = red[0];
    __syncthreads();

    float sum = 0.0f;
#pragma unroll
    for (int t = 0; t < 16; ++t) { v[t] = __expf(v[t] - mx); sum += v[t]; }
    red[tid] = sum;
    __syncthreads();
    for (int s = 128; s > 0; s >>= 1) {
        if (tid < s) red[tid] += red[tid + s];
        __syncthreads();
    }
    const float invZ = 1.0f / red[0];
    __syncthreads();

    float ss = 0.0f;
#pragma unroll
    for (int t = 0; t < 16; ++t) { v[t] *= invZ; ss += v[t] * v[t]; }
    red[tid] = ss;
    __syncthreads();
    for (int s = 128; s > 0; s >>= 1) {
        if (tid < s) red[tid] += red[tid + s];
        __syncthreads();
    }
    const float nrm = sqrtf(red[0]);
    const float sc  = sqrtf(curv[0]);
    const float n1  = fmaxf(nrm, 1e-5f);
    const float g   = tanhf(sc * n1) / (sc * n1);
    const float yn  = fmaxf(nrm * g, 1e-5f);
    const float mxn = (1.0f - 4e-3f) / sc;
    const float f   = (yn > mxn) ? (mxn / yn) : 1.0f;
    const float s2  = g * f;
#pragma unroll
    for (int t = 0; t < 16; ++t) p[tid + 256 * t] = v[t] * s2;
}

extern "C" void kernel_launch(void* const* d_in, const int* in_sizes, int n_in,
                              void* d_out, int out_size, void* d_ws, size_t ws_size,
                              hipStream_t stream) {
    const float* query   = (const float*)d_in[0];
    const float* context = (const float*)d_in[1];
    const float* W       = (const float*)d_in[2];
    const float* curv    = (const float*)d_in[3];

    // Shapes: W is d x d; query n x d; context m x d.
    const int d = (int)(sqrt((double)in_sizes[2]) + 0.5);   // 1024
    const int n = in_sizes[0] / d;                          // 4096
    const int m = in_sizes[1] / d;                          // 4096

    _Float16* Qh  = (_Float16*)d_ws;              // [n, d]  projected query  (f16)
    _Float16* Ch  = Qh + (size_t)n * d;           // [m, d]  projected context(f16)
    _Float16* Wh  = Ch + (size_t)m * d;           // [d, d]  W in f16
    _Float16* qWh = Wh + (size_t)d * d;           // [n, d]  (qProj @ W.T) in f16
    float*    Sout = (float*)d_out;               // [n, m]  scores -> attn

    // 1) expmap0 + project for query & context rows, emit f16
    prep_expmap_rows<<<n + m, 256, 0, stream>>>(query, context, Qh, Ch, curv, d, n);

    // 2) W -> f16
    cvt_f32_f16<<<(d * d + 255) / 256, 256, 0, stream>>>(W, Wh, d * d);

    // 3) GEMM1: qW[i,j] = sum_k Qh[i,k] * W[j,k]   ([n,d] = [n,d] x [d,d]^T)
    dim3 g1(d / 256, n / 128);
    gemm_nt_wmma<true><<<g1, 256, 0, stream>>>(Qh, Wh, nullptr, qWh, n, d, d);

    // 4) GEMM2: S[i,j] = sum_k qW[i,k] * ctx[j,k]  ([n,m] = [n,d] x [m,d]^T)
    dim3 g2(m / 256, n / 128);
    gemm_nt_wmma<false><<<g2, 256, 0, stream>>>(qWh, Ch, Sout, nullptr, n, m, d);

    // 5) softmax(S/d) + expmap0 + project, in place on d_out
    softmax_expmap_rows<<<n, 256, 0, stream>>>(Sout, curv, m, 1.0f / (float)d);
}
